// MambaFull_56856777065229
// MI455X (gfx1250) — matrix operand order repaired and checked by reading, not compiled, not run on hardware
//
#include <hip/hip_runtime.h>

// ---------------------------------------------------------------------------
// Mamba forward for MI455X (gfx1250, wave32).
// - All GEMMs: v_wmma_f32_16x16x32_bf16, TN layout (A[M,K], W[N,K], both
//   K-contiguous), per-lane fragment loads matching the ISA operand layouts.
// - Selective scan: B/C tile (201 x 32 f32, row stride 64) DMA'd once into
//   LDS via the Tensor Data Mover (tensor_load_to_lds + s_wait_tensorcnt),
//   then the 201-step recurrence runs barrier-free with state in registers.
// ---------------------------------------------------------------------------

typedef __attribute__((ext_vector_type(8)))  float        v8f;
typedef __attribute__((ext_vector_type(16))) __bf16       v16bf;
typedef __attribute__((ext_vector_type(4)))  unsigned int v4u;
typedef __attribute__((ext_vector_type(8)))  int          v8i;
typedef __attribute__((ext_vector_type(4)))  int          v4i;

struct BFragBits { v4u lo; v4u hi; };   // 32 bytes == v16bf

#define D_MODEL 512
#define D_INNER 1024
#define DT_RANK 32
#define D_STATE 16
#define SEQ_LEN 201
#define BATCH_N 64
#define M_TOK  (BATCH_N * SEQ_LEN)   // 12864, multiple of 64

__device__ __forceinline__ float silu_f(float v) {
    return v / (1.0f + __expf(-v));
}
__device__ __forceinline__ float softplus_f(float v) {
    return (v > 20.0f) ? v : log1pf(__expf(v));
}

// ---------------------------------------------------------------------------
// Generic bf16 WMMA GEMM: C[M,N] = A[M,K] * W[N,K]^T  (both K-contiguous).
// Block: 256 threads = 8 waves arranged 2(M) x 4(N); block tile 64 x 128,
// wave tile 32 x 32 (4 accumulators of 16x16 f32).
// mode 0: plain store; mode 1: +bias[col] then softplus (dt_proj).
// ---------------------------------------------------------------------------
__global__ __launch_bounds__(256) void gemm_bf16_wmma(
    const __bf16* __restrict__ A, const __bf16* __restrict__ W,
    float* __restrict__ C, int M, int N, int K,
    const float* __restrict__ bias, int mode)
{
    const int tid   = threadIdx.x;
    const int lane  = tid & 31;
    const int wave  = tid >> 5;
    const int nlane = lane & 15;
    const int khalf = lane >> 4;

    const int mBase = blockIdx.y * 64  + (wave >> 2) * 32;
    const int nBase = blockIdx.x * 128 + (wave & 3) * 32;

    v8f acc[2][2];
    #pragma unroll
    for (int i = 0; i < 2; ++i)
        #pragma unroll
        for (int j = 0; j < 2; ++j)
            acc[i][j] = v8f{};

    for (int k0 = 0; k0 < K; k0 += 32) {
        v16bf af[2], bf[2];
        // ---- A fragments (two 16-row tiles) ----
        #pragma unroll
        for (int ti = 0; ti < 2; ++ti) {
            int row = mBase + ti * 16 + nlane;
            if (row >= M) row = M - 1;
            const __bf16* p = A + (size_t)row * K + k0 + khalf * 8;
            BFragBits fb;
            fb.lo = *reinterpret_cast<const v4u*>(p);
            fb.hi = *reinterpret_cast<const v4u*>(p + 16);
            af[ti] = __builtin_bit_cast(v16bf, fb);
        }
        // ---- B fragments (two 16-col tiles; lane n = W row n) ----
        #pragma unroll
        for (int tj = 0; tj < 2; ++tj) {
            int row = nBase + tj * 16 + nlane;
            if (row >= N) row = N - 1;
            const __bf16* p = W + (size_t)row * K + k0 + khalf * 8;
            BFragBits fb;
            fb.lo = *reinterpret_cast<const v4u*>(p);
            fb.hi = *reinterpret_cast<const v4u*>(p + 16);
            bf[tj] = __builtin_bit_cast(v16bf, fb);
        }
        // Prefetch next K-slab of A (global_prefetch_b8 on gfx1250).
        if (k0 + 32 < K) {
            int row = mBase + nlane; if (row >= M) row = M - 1;
            __builtin_prefetch(A + (size_t)row * K + k0 + 32, 0, 1);
        }
        // ---- 4x v_wmma_f32_16x16x32_bf16 ----
        #pragma unroll
        for (int ti = 0; ti < 2; ++ti)
            #pragma unroll
            for (int tj = 0; tj < 2; ++tj)
                acc[ti][tj] = __builtin_amdgcn_wmma_f32_16x16x32_bf16(
                    false, af[ti], false, bf[tj],
                    (short)0, acc[ti][tj], false, false);
    }

    // ---- store: D layout is VGPR r -> row (r + 8*khalf), lane&15 -> col ----
    #pragma unroll
    for (int ti = 0; ti < 2; ++ti) {
        #pragma unroll
        for (int tj = 0; tj < 2; ++tj) {
            const int col = nBase + tj * 16 + nlane;
            #pragma unroll
            for (int r = 0; r < 8; ++r) {
                const int row = mBase + ti * 16 + r + 8 * khalf;
                if (row < M && col < N) {
                    float v = acc[ti][tj][r];
                    if (mode == 1) v = softplus_f(v + bias[col]);
                    C[(size_t)row * N + col] = v;
                }
            }
        }
    }
}

// ---------------------------------------------------------------------------
// Elementwise / fused kernels
// ---------------------------------------------------------------------------
__global__ __launch_bounds__(256) void embed_kernel(
    const float* __restrict__ x, const float* __restrict__ eW,
    const float* __restrict__ eb, float* __restrict__ h)
{
    int idx = blockIdx.x * 256 + threadIdx.x;          // over M_TOK * 512
    int d = idx & 511, tok = idx >> 9;
    h[(size_t)tok * 512 + d] =
        x[(size_t)tok * 2] * eW[d * 2] + x[(size_t)tok * 2 + 1] * eW[d * 2 + 1] + eb[d];
}

__global__ __launch_bounds__(256) void cvt_bf16_kernel(
    const float* __restrict__ in, __bf16* __restrict__ out, int n)
{
    int i = blockIdx.x * 256 + threadIdx.x;
    if (i < n) out[i] = (__bf16)in[i];
}

// residual accumulate + LayerNorm -> bf16 (one block per token, 512 features)
__global__ __launch_bounds__(256) void add_ln_kernel(
    const float* __restrict__ hin, float* __restrict__ res,
    __bf16* __restrict__ hn, const float* __restrict__ w,
    const float* __restrict__ b, int first)
{
    const int tok = blockIdx.x, tid = threadIdx.x;
    __shared__ float s1[256], s2[256];
    const float* hr = hin + (size_t)tok * 512;
    float*       rr = res + (size_t)tok * 512;
    float v0 = hr[tid], v1 = hr[tid + 256];
    if (!first) { v0 += rr[tid]; v1 += rr[tid + 256]; }
    rr[tid] = v0; rr[tid + 256] = v1;
    s1[tid] = v0 + v1; s2[tid] = v0 * v0 + v1 * v1;
    __syncthreads();
    for (int s = 128; s > 0; s >>= 1) {
        if (tid < s) { s1[tid] += s1[tid + s]; s2[tid] += s2[tid + s]; }
        __syncthreads();
    }
    const float mu = s1[0] * (1.0f / 512.0f);
    const float var = s2[0] * (1.0f / 512.0f) - mu * mu;
    const float rs = rsqrtf(var + 1e-5f);
    hn[(size_t)tok * 512 + tid]       = (__bf16)((v0 - mu) * rs * w[tid] + b[tid]);
    hn[(size_t)tok * 512 + tid + 256] = (__bf16)((v1 - mu) * rs * w[tid + 256] + b[tid + 256]);
}

// final residual + RMSNorm -> bf16
__global__ __launch_bounds__(256) void rms_kernel(
    const float* __restrict__ hin, const float* __restrict__ res,
    __bf16* __restrict__ xf, const float* __restrict__ w,
    const float* __restrict__ b)
{
    const int tok = blockIdx.x, tid = threadIdx.x;
    __shared__ float s2[256];
    float v0 = hin[(size_t)tok * 512 + tid]       + res[(size_t)tok * 512 + tid];
    float v1 = hin[(size_t)tok * 512 + tid + 256] + res[(size_t)tok * 512 + tid + 256];
    s2[tid] = v0 * v0 + v1 * v1;
    __syncthreads();
    for (int s = 128; s > 0; s >>= 1) {
        if (tid < s) s2[tid] += s2[tid + s];
        __syncthreads();
    }
    const float rs = rsqrtf(s2[0] * (1.0f / 512.0f) + 1e-5f);
    xf[(size_t)tok * 512 + tid]       = (__bf16)(v0 * rs * w[tid] + b[tid]);
    xf[(size_t)tok * 512 + tid + 256] = (__bf16)(v1 * rs * w[tid + 256] + b[tid + 256]);
}

// depthwise causal conv (width 4) + SiLU; writes f32 (for scan) and bf16 (GEMM A)
__global__ __launch_bounds__(256) void conv_silu_kernel(
    const float* __restrict__ xz, const float* __restrict__ cw,
    const float* __restrict__ cb, float* __restrict__ uf,
    __bf16* __restrict__ ub, int L)
{
    int idx = blockIdx.x * 256 + threadIdx.x;          // over M_TOK * 1024
    int c = idx & 1023, tok = idx >> 10;
    int bb = tok / L, l = tok - bb * L;
    float acc = cb[c];
    #pragma unroll
    for (int j = 0; j < 4; ++j) {
        int li = l - 3 + j;
        if (li >= 0) acc += xz[(size_t)(bb * L + li) * 2048 + c] * cw[c * 4 + j];
    }
    float s = silu_f(acc);
    uf[(size_t)tok * 1024 + c] = s;
    ub[(size_t)tok * 1024 + c] = (__bf16)s;
}

// extract dt_r (first 32 cols of x_dbl) -> compact bf16 [M,32]
__global__ __launch_bounds__(256) void dtr_extract_kernel(
    const float* __restrict__ xdbl, __bf16* __restrict__ dtr)
{
    int idx = blockIdx.x * 256 + threadIdx.x;          // over M_TOK * 32
    int r = idx & 31, tok = idx >> 5;
    dtr[(size_t)tok * 32 + r] = (__bf16)xdbl[(size_t)tok * 64 + r];
}

// ---------------------------------------------------------------------------
// Selective scan. One thread per (batch, channel); state[16] in registers.
// The whole (SEQ_LEN x 32) B/C slab for this batch is brought into LDS once
// by the Tensor Data Mover (2D tile, row stride 64 floats), so the 201-step
// recurrence runs with zero barriers and zero global loads for B/C.
// ---------------------------------------------------------------------------
__global__ __launch_bounds__(256) void scan_kernel(
    const float* __restrict__ dt, const float* __restrict__ uf,
    const float* __restrict__ xdbl, const float* __restrict__ A_log,
    float* __restrict__ ybuf, int L)
{
    const int bb = blockIdx.x;
    const int d  = blockIdx.y * 256 + threadIdx.x;

    __shared__ float bc[SEQ_LEN * 32];   // 25728 B of LDS

#if __has_builtin(__builtin_amdgcn_tensor_load_to_lds) && \
    __has_builtin(__builtin_amdgcn_s_wait_tensorcnt) && defined(__gfx1250__)
    if (threadIdx.x < 32) {              // wave 0 issues the TDM descriptor
        const float* gsrc = xdbl + (size_t)bb * L * 64 + 32;
        unsigned long long ga = (unsigned long long)(uintptr_t)gsrc;
        unsigned lds_off = (unsigned)(uintptr_t)(&bc[0]);   // low 32 bits = LDS addr
        // D# group 0: count=1 | lds_addr | global_addr[56:0] | type=2
        v4u g0 = { 1u, lds_off, (unsigned)ga,
                   (unsigned)((ga >> 32) & 0x01FFFFFFull) | 0x80000000u };
        // D# group 1: data_size=4B; tensor 32 x L (stride 64); tile 32 x L
        v8i g1 = { (int)(2u << 16),                 // wg_mask=0, data_size=2 (4B)
                   (int)(32u << 16),                // tensor_dim0 = 32 (bits 79:48 lo)
                   (int)(((unsigned)L & 0xFFFFu) << 16), // tensor_dim1 = L
                   (int)(32u << 16),                // tile_dim0 = 32
                   (int)(unsigned)L,                // tile_dim1 = L
                   64,                              // tensor_dim0_stride = 64
                   0, 0 };
        v4i gz = { 0, 0, 0, 0 };
#if __clang_major__ >= 23
        v8i gz8 = { 0, 0, 0, 0, 0, 0, 0, 0 };
        __builtin_amdgcn_tensor_load_to_lds(g0, g1, gz, gz, gz8, 0);
#else
        __builtin_amdgcn_tensor_load_to_lds(g0, g1, gz, gz, 0);
#endif
        __builtin_amdgcn_s_wait_tensorcnt(0);
    }
#else
    // Fallback: cooperative load of the B/C slab.
    for (int idx = threadIdx.x; idx < L * 32; idx += 256)
        bc[idx] = xdbl[((size_t)bb * L + (idx >> 5)) * 64 + 32 + (idx & 31)];
#endif
    __syncthreads();

    float Av[16], st[16];
    #pragma unroll
    for (int n = 0; n < 16; ++n) {
        Av[n] = -__expf(A_log[(size_t)d * 16 + n]);
        st[n] = 0.0f;
    }
    for (int t = 0; t < L; ++t) {
        const size_t tok = (size_t)bb * L + t;
        const float dtv = dt[tok * 1024 + d];
        const float uv  = uf[tok * 1024 + d];
        const float du  = dtv * uv;
        const float* bcr = &bc[t * 32];
        float y = 0.0f;
        #pragma unroll
        for (int n = 0; n < 16; ++n) {
            st[n] = __expf(dtv * Av[n]) * st[n] + du * bcr[n];
            y += st[n] * bcr[16 + n];
        }
        ybuf[tok * 1024 + d] = y;
    }
}

// y = (scan + Dskip*u) * silu(z)  -> bf16 (A operand of out_proj)
__global__ __launch_bounds__(256) void gate_kernel(
    const float* __restrict__ ybuf, const float* __restrict__ uf,
    const float* __restrict__ xz, const float* __restrict__ Dv,
    __bf16* __restrict__ yb)
{
    int idx = blockIdx.x * 256 + threadIdx.x;          // over M_TOK * 1024
    int c = idx & 1023, tok = idx >> 10;
    float yv = ybuf[(size_t)tok * 1024 + c] + Dv[c] * uf[(size_t)tok * 1024 + c];
    float z  = xz[(size_t)tok * 2048 + 1024 + c];
    yb[(size_t)tok * 1024 + c] = (__bf16)(yv * silu_f(z));
}

// ---------------------------------------------------------------------------
extern "C" void kernel_launch(void* const* d_in, const int* in_sizes, int n_in,
                              void* d_out, int out_size, void* d_ws, size_t ws_size,
                              hipStream_t stream) {
    const float* x        = (const float*)d_in[0];
    const float* embed_W  = (const float*)d_in[2];
    const float* embed_b  = (const float*)d_in[3];
    const float* norm_w   = (const float*)d_in[4];
    const float* norm_b   = (const float*)d_in[5];
    const float* in_W     = (const float*)d_in[6];
    const float* conv_w   = (const float*)d_in[7];
    const float* conv_b   = (const float*)d_in[8];
    const float* xproj_W  = (const float*)d_in[9];
    const float* dtproj_W = (const float*)d_in[10];
    const float* dtproj_b = (const float*)d_in[11];
    const float* A_log    = (const float*)d_in[12];
    const float* Dskip    = (const float*)d_in[13];
    const float* out_W    = (const float*)d_in[14];
    const float* normf_w  = (const float*)d_in[15];
    const float* normf_b  = (const float*)d_in[16];
    const float* head_W   = (const float*)d_in[17];
    float* out = (float*)d_out;

    const int L = SEQ_LEN, M = M_TOK;

    // bump allocator over workspace
    char* wp = (char*)d_ws;
    auto alloc = [&](size_t bytes) -> void* {
        void* r = (void*)wp;
        wp += (bytes + 255) & ~(size_t)255;
        return r;
    };
    float*  res    = (float*)alloc((size_t)M * 512 * 4);
    float*  h      = (float*)alloc((size_t)M * 512 * 4);
    float*  xz     = (float*)alloc((size_t)M * 2048 * 4);
    float*  u_f    = (float*)alloc((size_t)M * 1024 * 4);
    float*  xdbl   = (float*)alloc((size_t)M * 64 * 4);
    float*  dtbuf  = (float*)alloc((size_t)M * 1024 * 4);
    float*  ybuf   = (float*)alloc((size_t)M * 1024 * 4);
    __bf16* hn_b   = (__bf16*)alloc((size_t)M * 512 * 2);   // also reused as xf
    __bf16* u_b    = (__bf16*)alloc((size_t)M * 1024 * 2);
    __bf16* dtr_b  = (__bf16*)alloc((size_t)M * 32 * 2);
    __bf16* y_b    = (__bf16*)alloc((size_t)M * 1024 * 2);
    __bf16* win_b  = (__bf16*)alloc((size_t)2048 * 512 * 2);
    __bf16* wxp_b  = (__bf16*)alloc((size_t)64 * 1024 * 2);
    __bf16* wdt_b  = (__bf16*)alloc((size_t)1024 * 32 * 2);
    __bf16* wout_b = (__bf16*)alloc((size_t)512 * 1024 * 2);
    __bf16* whd_b  = (__bf16*)alloc((size_t)200 * 512 * 2);

    auto cvt = [&](const float* src, __bf16* dst, int n) {
        cvt_bf16_kernel<<<(n + 255) / 256, 256, 0, stream>>>(src, dst, n);
    };
    auto gemm = [&](const __bf16* Ap, const __bf16* Wp, float* Cp,
                    int Mi, int Ni, int Ki, const float* bias, int mode) {
        dim3 g((Ni + 127) / 128, (Mi + 63) / 64);
        gemm_bf16_wmma<<<g, 256, 0, stream>>>(Ap, Wp, Cp, Mi, Ni, Ki, bias, mode);
    };

    embed_kernel<<<(M * 512) / 256, 256, 0, stream>>>(x, embed_W, embed_b, h);

    for (int i = 0; i < 3; ++i) {
        cvt(in_W     + (size_t)i * 2048 * 512, win_b,  2048 * 512);
        cvt(xproj_W  + (size_t)i * 64 * 1024,  wxp_b,  64 * 1024);
        cvt(dtproj_W + (size_t)i * 1024 * 32,  wdt_b,  1024 * 32);
        cvt(out_W    + (size_t)i * 512 * 1024, wout_b, 512 * 1024);

        add_ln_kernel<<<M, 256, 0, stream>>>(h, res, hn_b,
                                             norm_w + i * 512, norm_b + i * 512, i == 0);
        gemm(hn_b, win_b, xz, M, 2048, 512, nullptr, 0);
        conv_silu_kernel<<<(M * 1024) / 256, 256, 0, stream>>>(
            xz, conv_w + (size_t)i * 1024 * 4, conv_b + i * 1024, u_f, u_b, L);
        gemm(u_b, wxp_b, xdbl, M, 64, 1024, nullptr, 0);
        dtr_extract_kernel<<<(M * 32) / 256, 256, 0, stream>>>(xdbl, dtr_b);
        gemm(dtr_b, wdt_b, dtbuf, M, 1024, 32, dtproj_b + i * 1024, 1);
        scan_kernel<<<dim3(BATCH_N, 4), 256, 0, stream>>>(
            dtbuf, u_f, xdbl, A_log + (size_t)i * 1024 * 16, ybuf, L);
        gate_kernel<<<(M * 1024) / 256, 256, 0, stream>>>(
            ybuf, u_f, xz, Dskip + i * 1024, y_b);
        gemm(y_b, wout_b, h, M, 512, 1024, nullptr, 0);
    }

    rms_kernel<<<M, 256, 0, stream>>>(h, res, hn_b, normf_w, normf_b);
    cvt(head_W, whd_b, 200 * 512);
    gemm(hn_b, whd_b, out, M, 200, 512, nullptr, 0);
}
